// MarkovAttnBlock_71330816852022
// MI455X (gfx1250) — compile-verified, matrix operand order checked by reading, not run on hardware
//
#include <hip/hip_runtime.h>
#include <hip/hip_bf16.h>

// ---------------------------------------------------------------------------
// MarkovAttnBlock on gfx1250 (CDNA5, wave32, WMMA 16x16x32 f16)
//
// Pipeline:
//   0) zero workspace (padded buffers rely on zero borders)
//   1) convert weights to f16 (struct/feat 1x1, pre 3x3 padded to 448 ch)
//   2) f_t = struct_w * target   (WMMA GEMM, f16 out, NHWC)
//      f_s = feat_w  * source    (WMMA GEMM, f16 out, NHWC, 20/20 spatial pad)
//   3) correlation cost volume   (WMMA banded GEMM + LDS band extraction)
//   4) pre 3x3 conv 441->128     (WMMA; corr rows staged to LDS with the
//                                 Tensor Data Mover, TENSORcnt fenced)
//   5) flow/offset/mask heads    (scalar conv, async-to-LDS staged weights)
//   6) markov deformable attention (softmax + bilinear gather)
// ---------------------------------------------------------------------------

typedef __attribute__((ext_vector_type(16))) _Float16 v16h;
typedef __attribute__((ext_vector_type(8)))  float    v8f;
typedef __attribute__((ext_vector_type(8)))  unsigned int v8u;
typedef __attribute__((ext_vector_type(4)))  unsigned int v4u;

#define BATCH 8
#define CIN   256
#define FC    128
#define HH    64
#define WW    64
#define ND    21
#define C441  441
#define C448  448
#define HP    104   // f_s vertical pad:  64 + 2*20
#define WP    112   // f_s horizontal pad: 64 + 2*20 + 8 slack for 64-wide GEMM window
#define CH    66    // corr/hidden spatially padded (pad=1)

// ---- workspace layout (bytes) ----------------------------------------------
static constexpr size_t WS_FT   = 0;
static constexpr size_t SZ_FT   = (size_t)BATCH * HH * WW * FC * 2;
static constexpr size_t WS_FSP  = WS_FT + SZ_FT;
static constexpr size_t SZ_FSP  = (size_t)BATCH * HP * WP * FC * 2;
static constexpr size_t WS_CORR = WS_FSP + SZ_FSP;
static constexpr size_t SZ_CORR = (size_t)BATCH * CH * CH * C448 * 2;
static constexpr size_t WS_HID  = WS_CORR + SZ_CORR;
static constexpr size_t SZ_HID  = (size_t)BATCH * CH * CH * FC * 4;
static constexpr size_t WS_SWH  = WS_HID + SZ_HID;
static constexpr size_t SZ_WH   = (size_t)FC * CIN * 2;
static constexpr size_t WS_FWH  = WS_SWH + SZ_WH;
static constexpr size_t WS_PWH  = WS_FWH + SZ_WH;
static constexpr size_t SZ_PWH  = (size_t)9 * FC * C448 * 2;
static constexpr size_t WS_TOTAL = WS_PWH + SZ_PWH;   // ~82.5 MB

// ---- output layout (float elements) ----------------------------------------
static constexpr size_t O_OUT  = 0;
static constexpr size_t O_FLOW = (size_t)BATCH * CIN * HH * WW;
static constexpr size_t O_OFF  = O_FLOW + (size_t)BATCH * 2 * HH * WW;
static constexpr size_t O_MASK = O_OFF + (size_t)BATCH * 18 * HH * WW;

// ---------------------------------------------------------------------------
// WMMA helpers (CDNA5 16-bit fragment striping, ISA 7.12.2)
// ---------------------------------------------------------------------------
__device__ __forceinline__ v8f wmma_f16(v16h a, v16h b, v8f c) {
  return __builtin_amdgcn_wmma_f32_16x16x32_f16(
      false, a, false, b, (short)0, c, false, false);
}

template <typename PT>
__device__ __forceinline__ v16h load_frag(PT row_base, int stride, int lane) {
  const int row   = lane & 15;
  const int khalf = (lane >> 4) << 3;
  PT r = row_base + (long)row * stride;
  v8u tmp;
#pragma unroll
  for (int p = 0; p < 8; ++p) {
    const int k = ((p & 4) ? 16 : 0) + khalf + ((p & 3) << 1);
    tmp[p] = *(const unsigned int*)(&r[k]);   // dword = 2 packed f16
  }
  return __builtin_bit_cast(v16h, tmp);
}

__device__ __forceinline__ float leaky(float x) {
  return x >= 0.0f ? x : 0.1f * x;
}

// ---------------------------------------------------------------------------
// CDNA5 async / tensor data-movement helpers.
// LDS destination = wave-relative byte offset = low 32 bits of generic pointer
// (LDS aperture keeps the offset in addr[31:0]).
// ---------------------------------------------------------------------------
__device__ __forceinline__ unsigned lds_off_of(const void* p) {
  return (unsigned)(unsigned long long)p;
}
__device__ __forceinline__ void async_copy_b32(unsigned lds_off, const void* gaddr) {
  asm volatile("global_load_async_to_lds_b32 %0, %1, off"
               :: "v"(lds_off), "v"((unsigned long long)gaddr) : "memory");
}
__device__ __forceinline__ void wait_async0() {
  asm volatile("s_wait_asynccnt 0x0" ::: "memory");
}

// TDM: DMA a contiguous run of nelem8 x 8-byte elements from global to LDS.
// Descriptor per ISA 8.3/8.4: group0 = {count=1, lds_addr, global_addr, type=2},
// group1 = {data_size=8B, tensor_dim0 = tile_dim0 = nelem8, tensor_dim1 = 1,
//           tile_dim1 = 1, tensor_dim0_stride = nelem8}. workgroup_mask = 0
// (not in a cluster). Issued once per wave; EXEC is ignored by TDM.
__device__ __forceinline__ void tdm_copy_to_lds_8B(unsigned lds_off,
                                                   const void* gsrc,
                                                   unsigned nelem8) {
  const unsigned long long ga = (unsigned long long)gsrc;
  v4u g0;
  g0[0] = 1u;                                               // count=1, user D#
  g0[1] = lds_off;                                          // lds_addr[31:0]
  g0[2] = (unsigned)ga;                                     // global_addr[31:0]
  g0[3] = ((unsigned)(ga >> 32) & 0x01FFFFFFu) | (2u << 30);// addr[56:32]|type=2
  v8u g1;
  g1[0] = 3u << 16;                      // wg_mask=0, data_size=3 (8 bytes)
  g1[1] = (nelem8 & 0xFFFFu) << 16;      // tensor_dim0[15:0]  (bits 63:48)
  g1[2] = (nelem8 >> 16) | (1u << 16);   // tensor_dim0[31:16] | tensor_dim1=1
  g1[3] = (nelem8 & 0xFFFFu) << 16;      // tile_dim0          (bits 127:112)
  g1[4] = 1u;                            // tile_dim1 = 1
  g1[5] = nelem8;                        // tensor_dim0_stride[31:0]
  g1[6] = 0u;
  g1[7] = 0u;
  asm volatile("tensor_load_to_lds %0, %1" :: "s"(g0), "s"(g1) : "memory");
}

// ---------------------------------------------------------------------------
// 0) zero workspace
// ---------------------------------------------------------------------------
__global__ __launch_bounds__(256) void zero_ws_kernel(uint4* p, long n16) {
  long i = (long)blockIdx.x * blockDim.x + threadIdx.x;
  long stride = (long)gridDim.x * blockDim.x;
  for (; i < n16; i += stride) p[i] = make_uint4(0u, 0u, 0u, 0u);
}

// ---------------------------------------------------------------------------
// 1) weight conversion to f16 (pre_w re-layout OIHW -> [tap][oc][448])
// ---------------------------------------------------------------------------
__global__ __launch_bounds__(256) void cvt_weights_kernel(
    const float* __restrict__ sw, const float* __restrict__ fw,
    const float* __restrict__ pw,
    _Float16* __restrict__ swh, _Float16* __restrict__ fwh,
    _Float16* __restrict__ pwh) {
  const int i = blockIdx.x * 256 + threadIdx.x;
  if (i < FC * CIN) {
    swh[i] = (_Float16)sw[i];
    fwh[i] = (_Float16)fw[i];
  }
  const int total = FC * C441 * 9;
  for (int id = i; id < total; id += gridDim.x * 256) {
    const int oc  = id / (C441 * 9);
    const int rem = id - oc * (C441 * 9);
    const int c   = rem / 9;
    const int tap = rem - c * 9;
    pwh[((long)tap * FC + oc) * C448 + c] = (_Float16)pw[id];
  }
}

// ---------------------------------------------------------------------------
// 2) 1x1 conv as WMMA GEMM
// ---------------------------------------------------------------------------
__global__ __launch_bounds__(256) void conv1x1_wmma_kernel(
    const float* __restrict__ in,     // [8][256][64][64] NCHW f32
    const _Float16* __restrict__ Wh,  // [128][256] f16
    const float* __restrict__ bias,   // [128]
    _Float16* __restrict__ out, int ow, int oy0, int ox0, long obstride) {
  __shared__ __align__(16) _Float16 tile[16 * 264];   // [px][c]
  const int tid = threadIdx.x;
  const int blk = blockIdx.x;           // 8*64*4
  const int xt  = blk & 3;
  const int y   = (blk >> 2) & 63;
  const int b   = blk >> 8;
  const int x0  = xt * 16;

  {  // stage activations: NCHW f32 -> LDS [px][c] f16 (conversion => VALU path)
    const int px = tid & 15;
    const int c0 = tid >> 4;
    const float* gp = in + (long)b * CIN * HH * WW + (long)y * WW + x0 + px;
    for (int c = c0; c < CIN; c += 16)
      tile[px * 264 + c] = (_Float16)gp[(long)c * HH * WW];
  }
  __syncthreads();

  const int wave = tid >> 5, lane = tid & 31;
  v8f acc;
#pragma unroll
  for (int i = 0; i < 8; ++i) acc[i] = 0.0f;

#pragma unroll 1
  for (int kc = 0; kc < CIN / 32; ++kc) {
    v16h a  = load_frag(Wh + (long)wave * 16 * CIN + kc * 32, CIN, lane);
    v16h bf = load_frag((const _Float16*)tile + kc * 32, 264, lane);
    acc = wmma_f16(a, bf, acc);
  }

  _Float16* ob = out + (long)b * obstride;
#pragma unroll
  for (int v = 0; v < 8; ++v) {
    const int oc = wave * 16 + v + ((lane >> 4) << 3);
    const int n  = lane & 15;
    const float val = acc[v] + bias[oc];
    ob[((long)(y + oy0) * ow + (x0 + n + ox0)) * FC + oc] = (_Float16)val;
  }
}

// ---------------------------------------------------------------------------
// 3) correlation: banded GEMM per (b, y, dyi, xtile) wave-task
// ---------------------------------------------------------------------------
__global__ __launch_bounds__(256) void corr_wmma_kernel(
    const _Float16* __restrict__ ft,   // [8][64][64][128]
    const _Float16* __restrict__ fsp,  // [8][104][112][128]
    _Float16* __restrict__ corr) {     // [8][66][66][448]
  __shared__ float cbuf[8 * 16 * 64];
  const int tid = threadIdx.x, wave = tid >> 5, lane = tid & 31;
  const int t  = blockIdx.x * 8 + wave;      // 8*64*21*4 tasks
  const int xt = t & 3;
  int r = t >> 2;
  const int dyi = r % ND;  r /= ND;
  const int y   = r & 63;
  const int b   = r >> 6;
  const int x0  = xt * 16;

  const _Float16* Ab = ft  + (((long)b * HH + y) * WW + x0) * FC;
  const _Float16* Bb = fsp + (((long)b * HP + (y + 2 * dyi)) * WP + x0) * FC;

  v8f a0, a1, a2, a3;
#pragma unroll
  for (int i = 0; i < 8; ++i) { a0[i] = 0.f; a1[i] = 0.f; a2[i] = 0.f; a3[i] = 0.f; }

#pragma unroll 1
  for (int kc = 0; kc < FC / 32; ++kc) {
    v16h af = load_frag(Ab + kc * 32, FC, lane);
    v16h b0 = load_frag(Bb + kc * 32, FC, lane);
    a0 = wmma_f16(af, b0, a0);
    v16h b1 = load_frag(Bb + (long)16 * FC + kc * 32, FC, lane);
    a1 = wmma_f16(af, b1, a1);
    v16h b2 = load_frag(Bb + (long)32 * FC + kc * 32, FC, lane);
    a2 = wmma_f16(af, b2, a2);
    v16h b3 = load_frag(Bb + (long)48 * FC + kc * 32, FC, lane);
    a3 = wmma_f16(af, b3, a3);
  }

  float* w = cbuf + wave * 1024;
  const int hi = (lane >> 4) << 3, n = lane & 15;
#pragma unroll
  for (int v = 0; v < 8; ++v) {
    const int m = v + hi;
    w[m * 64 +  0 + n] = a0[v];
    w[m * 64 + 16 + n] = a1[v];
    w[m * 64 + 32 + n] = a2[v];
    w[m * 64 + 48 + n] = a3[v];
  }

  // band extraction: corr[x0+m, dxi] = C[m, m+2*dxi] / 128, leaky
  _Float16* cp = corr + ((long)b * CH + (y + 1)) * CH * C448;
#pragma unroll 1
  for (int it = 0; it < 11; ++it) {
    const int idx = it * 32 + lane;
    if (idx < 336) {
      const int m = idx / ND, dxi = idx - m * ND;
      float val = w[m * 64 + m + 2 * dxi] * (1.0f / 128.0f);
      val = leaky(val);
      cp[(long)(x0 + m + 1) * C448 + (dyi * ND + dxi)] = (_Float16)val;
    }
  }
}

// ---------------------------------------------------------------------------
// 4) pre 3x3 conv 441->128 via WMMA. Block = (b,y): each needed corr row
//    (66 x 448 f16 = 59,136 B) is DMA'd into LDS by the Tensor Data Mover
//    (one tensor_load_to_lds per row, TENSORcnt fenced), then 8 waves x 4
//    pixel-tiles consume it for the 3 taps sharing ky.
// ---------------------------------------------------------------------------
__global__ __launch_bounds__(256) void preconv_wmma_kernel(
    const _Float16* __restrict__ corr,  // [8][66][66][448]
    const _Float16* __restrict__ pwh,   // [9][128][448]
    const float* __restrict__ pb,       // [128]
    float* __restrict__ hid) {          // [8][66][66][128]
  __shared__ __align__(16) _Float16 rowbuf[CH * C448];   // 59136 B
  const int tid = threadIdx.x, wave = tid >> 5, lane = tid & 31;
  const int blk = blockIdx.x;           // 8*64
  const int y = blk & 63, b = blk >> 6;
  const unsigned lds_base = lds_off_of(rowbuf);

  v8f acc[4];
#pragma unroll
  for (int p = 0; p < 4; ++p)
#pragma unroll
    for (int i = 0; i < 8; ++i) acc[p][i] = 0.0f;

#pragma unroll 1
  for (int ky = 0; ky < 3; ++ky) {
    __syncthreads();   // previous row fully consumed (ds reads done pre-wmma)
    if (wave == 0) {   // one TDM DMA per block per row
      const char* grow =
          (const char*)(corr + ((long)b * CH + (y + ky)) * CH * C448);
      tdm_copy_to_lds_8B(lds_base, grow, (CH * C448 * 2) / 8);   // 7392 x 8B
      __builtin_amdgcn_s_wait_tensorcnt(0);
    }
    __syncthreads();

#pragma unroll 1
    for (int kx = 0; kx < 3; ++kx) {
      const int tap = ky * 3 + kx;
      const _Float16* At = pwh + ((long)tap * FC + wave * 16) * C448;
#pragma unroll 1
      for (int kc = 0; kc < C448 / 32; ++kc) {
        v16h a = load_frag(At + kc * 32, C448, lane);
#pragma unroll
        for (int pt = 0; pt < 4; ++pt) {
          v16h bf = load_frag((const _Float16*)rowbuf +
                                  (long)(pt * 16 + kx) * C448 + kc * 32,
                              C448, lane);
          acc[pt] = wmma_f16(a, bf, acc[pt]);
        }
      }
    }
  }

  float* hb = hid + ((long)b * CH + (y + 1)) * CH * FC;
  const int hi = (lane >> 4) << 3, n = lane & 15;
#pragma unroll
  for (int pt = 0; pt < 4; ++pt)
#pragma unroll
    for (int v = 0; v < 8; ++v) {
      const int oc = wave * 16 + v + hi;
      const int x  = pt * 16 + n;
      hb[(long)(x + 1) * FC + oc] = leaky(acc[pt][v] + pb[oc]);
    }
}

// ---------------------------------------------------------------------------
// 5) flow/offset/mask heads: per-pixel 29-ch 3x3 conv over hidden.
//    Per-tap weight block staged to LDS with async b32 copies.
// ---------------------------------------------------------------------------
__global__ __launch_bounds__(256) void heads_kernel(
    const float* __restrict__ hid,  // [8][66][66][128]
    const float* __restrict__ fw, const float* __restrict__ fb,
    const float* __restrict__ ofw, const float* __restrict__ ob,
    const float* __restrict__ mw, const float* __restrict__ mb,
    float* __restrict__ out) {
  __shared__ __align__(16) float wtap[29 * 128];
  const int tid = threadIdx.x;
  const int id = blockIdx.x * 256 + tid;      // 32768 pixels
  const int x = id & 63, y = (id >> 6) & 63, b = id >> 12;
  const unsigned wl = lds_off_of(wtap);

  float acc[29];
#pragma unroll
  for (int i = 0; i < 29; ++i) acc[i] = 0.0f;

  for (int tap = 0; tap < 9; ++tap) {
    __syncthreads();
    for (int j = tid; j < 29 * 128; j += 256) {
      const int oc = j >> 7, c = j & 127;
      const float* src;
      if (oc < 2)       src = fw + oc * 1152 + c * 9 + tap;
      else if (oc < 20) src = ofw + (oc - 2) * 1152 + c * 9 + tap;
      else              src = mw + (oc - 20) * 1152 + c * 9 + tap;
      async_copy_b32(wl + (unsigned)j * 4u, src);
    }
    wait_async0();
    __syncthreads();
    const float* hp = hid + (((long)b * CH + (y + tap / 3)) * CH + (x + tap % 3)) * FC;
    for (int c = 0; c < FC; ++c) {
      const float h = hp[c];
#pragma unroll
      for (int oc = 0; oc < 29; ++oc) acc[oc] += h * wtap[oc * 128 + c];
    }
  }

  const long pix = (long)y * WW + x;
#pragma unroll
  for (int oc = 0; oc < 2; ++oc)
    out[O_FLOW + ((long)b * 2 + oc) * (HH * WW) + pix] = acc[oc] + fb[oc];
#pragma unroll
  for (int oc = 0; oc < 18; ++oc)
    out[O_OFF + ((long)b * 18 + oc) * (HH * WW) + pix] = acc[2 + oc] + ob[oc];
#pragma unroll
  for (int oc = 0; oc < 9; ++oc)
    out[O_MASK + ((long)b * 9 + oc) * (HH * WW) + pix] = leaky(acc[20 + oc] + mb[oc]);
}

// ---------------------------------------------------------------------------
// 6) markov deformable attention: 64 threads per pixel, 4 channels each.
// ---------------------------------------------------------------------------
__global__ __launch_bounds__(256) void markov_kernel(
    const float* __restrict__ src,     // [8][256][64][64]
    const float* __restrict__ aux,     // d_out (flow/off/mask from heads)
    float* __restrict__ outp) {
  const int tid = threadIdx.x;
  const int pid = blockIdx.x * 4 + (tid >> 6);  // 32768 pixels
  const int sub = tid & 63;
  const int x = pid & 63, y = (pid >> 6) & 63, b = pid >> 12;
  const long pix = (long)y * WW + x;

  float wk[9];
  {
    const float* mp = aux + O_MASK + (long)b * 9 * (HH * WW) + pix;
    float mx = -3.0e38f;
#pragma unroll
    for (int k = 0; k < 9; ++k) { wk[k] = mp[(long)k * (HH * WW)]; mx = fmaxf(mx, wk[k]); }
    float s = 0.0f;
#pragma unroll
    for (int k = 0; k < 9; ++k) { wk[k] = __expf(wk[k] - mx); s += wk[k]; }
    const float inv = 1.0f / s;
#pragma unroll
    for (int k = 0; k < 9; ++k) wk[k] *= inv;
  }

  const float fx = aux[O_FLOW + (long)b * 2 * (HH * WW) + pix];
  const float fy = aux[O_FLOW + (long)b * 2 * (HH * WW) + (HH * WW) + pix];
  const float* op = aux + O_OFF + (long)b * 18 * (HH * WW) + pix;
  const float* sb = src + (long)b * CIN * (HH * WW);

  float a0 = 0.f, a1 = 0.f, a2 = 0.f, a3 = 0.f;
#pragma unroll 1
  for (int k = 0; k < 9; ++k) {
    const float offy = op[(long)(k * 2 + 0) * (HH * WW)];
    const float offx = op[(long)(k * 2 + 1) * (HH * WW)];
    const float py = (float)y + fy + (float)(k / 3 - 1) + offy;
    const float px = (float)x + fx + (float)(k % 3 - 1) + offx;
    const float y0f = floorf(py), x0f = floorf(px);
    const float wy1 = py - y0f, wy0 = 1.0f - wy1;
    const float wx1 = px - x0f, wx0 = 1.0f - wx1;
    const int y0i = (int)y0f, x0i = (int)x0f;
    const float sw = wk[k];
#pragma unroll
    for (int cy = 0; cy < 2; ++cy)
#pragma unroll
      for (int cx = 0; cx < 2; ++cx) {
        const int yy = y0i + cy, xx = x0i + cx;
        if (yy >= 0 && yy < HH && xx >= 0 && xx < WW) {
          const float cw = sw * (cy ? wy1 : wy0) * (cx ? wx1 : wx0);
          const float* g = sb + (long)yy * WW + xx;
          a0 += cw * g[(long)(sub)       * (HH * WW)];
          a1 += cw * g[(long)(sub + 64)  * (HH * WW)];
          a2 += cw * g[(long)(sub + 128) * (HH * WW)];
          a3 += cw * g[(long)(sub + 192) * (HH * WW)];
        }
      }
  }
  outp[O_OUT + ((long)b * CIN + sub)       * (HH * WW) + pix] = a0;
  outp[O_OUT + ((long)b * CIN + sub + 64)  * (HH * WW) + pix] = a1;
  outp[O_OUT + ((long)b * CIN + sub + 128) * (HH * WW) + pix] = a2;
  outp[O_OUT + ((long)b * CIN + sub + 192) * (HH * WW) + pix] = a3;
}

// ---------------------------------------------------------------------------
extern "C" void kernel_launch(void* const* d_in, const int* in_sizes, int n_in,
                              void* d_out, int out_size, void* d_ws, size_t ws_size,
                              hipStream_t stream) {
  (void)in_sizes; (void)n_in; (void)out_size;
  if (ws_size < WS_TOTAL) return;

  const float* source   = (const float*)d_in[0];
  const float* target   = (const float*)d_in[1];
  const float* feat_w   = (const float*)d_in[2];
  const float* feat_b   = (const float*)d_in[3];
  const float* struct_w = (const float*)d_in[4];
  const float* struct_b = (const float*)d_in[5];
  const float* pre_w    = (const float*)d_in[6];
  const float* pre_b    = (const float*)d_in[7];
  const float* flow_w   = (const float*)d_in[8];
  const float* flow_b   = (const float*)d_in[9];
  const float* off_w    = (const float*)d_in[10];
  const float* off_b    = (const float*)d_in[11];
  const float* mask_w   = (const float*)d_in[12];
  const float* mask_b   = (const float*)d_in[13];

  char* ws = (char*)d_ws;
  _Float16* ftb  = (_Float16*)(ws + WS_FT);
  _Float16* fsp  = (_Float16*)(ws + WS_FSP);
  _Float16* corr = (_Float16*)(ws + WS_CORR);
  float*    hid  = (float*)(ws + WS_HID);
  _Float16* swh  = (_Float16*)(ws + WS_SWH);
  _Float16* fwh  = (_Float16*)(ws + WS_FWH);
  _Float16* pwh  = (_Float16*)(ws + WS_PWH);
  float* out = (float*)d_out;

  zero_ws_kernel<<<2048, 256, 0, stream>>>((uint4*)d_ws, (long)(WS_TOTAL / 16));
  cvt_weights_kernel<<<1985, 256, 0, stream>>>(struct_w, feat_w, pre_w, swh, fwh, pwh);

  conv1x1_wmma_kernel<<<BATCH * 64 * 4, 256, 0, stream>>>(
      target, swh, struct_b, ftb, WW, 0, 0, (long)HH * WW * FC);
  conv1x1_wmma_kernel<<<BATCH * 64 * 4, 256, 0, stream>>>(
      source, fwh, feat_b, fsp, WP, 20, 20, (long)HP * WP * FC);

  corr_wmma_kernel<<<(BATCH * 64 * ND * 4) / 8, 256, 0, stream>>>(ftb, fsp, corr);
  preconv_wmma_kernel<<<BATCH * 64, 256, 0, stream>>>(corr, pwh, pre_b, hid);
  heads_kernel<<<(BATCH * HH * WW) / 256, 256, 0, stream>>>(
      hid, flow_w, flow_b, off_w, off_b, mask_w, mask_b, out);
  markov_kernel<<<(BATCH * HH * WW) / 4, 256, 0, stream>>>(source, out, out);
}
